// MultiHeadSelfAttention_22617297781224
// MI455X (gfx1250) — compile-verified, matrix operand order checked by reading, not run on hardware
//
#include <hip/hip_runtime.h>

// ---------------------------------------------------------------------------
// MI455X (gfx1250) multi-head self-attention, bf16 WMMA with f32 accumulate.
// GEMMs: async global->LDS double-buffered pipeline feeding v_wmma.
// ---------------------------------------------------------------------------

typedef __attribute__((ext_vector_type(16))) __bf16        v16bf;
typedef __attribute__((ext_vector_type(8)))  float         v8f;
typedef __attribute__((ext_vector_type(8)))  unsigned int  v8u;
typedef __attribute__((ext_vector_type(4)))  int           v4i;

#define BATCH 4
#define SEQ   2048
#define EMB   1024
#define NH    16
#define HD    64
#define EMB3  3072

// ---- CDNA5 async global->LDS path (guarded; fallback = plain staging) -----
#if defined(__has_builtin)
#if __has_builtin(__builtin_amdgcn_global_load_async_to_lds_b128)
#define USE_ASYNC_LDS 1
#endif
#endif

#if defined(USE_ASYNC_LDS)
#if __has_builtin(__builtin_amdgcn_s_wait_asynccnt)
#define ASYNC_WAIT(n) __builtin_amdgcn_s_wait_asynccnt(n)
#else
#define ASYNC_WAIT(n) __asm__ volatile("s_wait_asynccnt " #n ::: "memory")
#endif
#else
#define ASYNC_WAIT(n)
#endif

__device__ __forceinline__ void async_copy16(const unsigned short* g, unsigned short* l) {
#if defined(USE_ASYNC_LDS)
  __builtin_amdgcn_global_load_async_to_lds_b128(
      (__attribute__((address_space(1))) v4i*)g,     // AS1, const cast away
      (__attribute__((address_space(3))) v4i*)l, 0, 0);
#else
  *(uint4*)l = *(const uint4*)g;
#endif
}

__device__ __forceinline__ unsigned short f2bf(float f) {
  unsigned int u = __builtin_bit_cast(unsigned int, f);
  u += 0x7FFFu + ((u >> 16) & 1u);          // round-to-nearest-even
  return (unsigned short)(u >> 16);
}

__device__ __forceinline__ v8f vzero8() {
  v8f z;
#pragma unroll
  for (int i = 0; i < 8; ++i) z[i] = 0.0f;
  return z;
}

// Load a 16x32 bf16 fragment in the CDNA5 WMMA 16-bit A/B register layout:
//   lane (0..15 | 16..31) holds row `row + (lane&15)`,
//   VGPR v (pair-packed) holds K = (v<4?0:16) + 2*(v&3) + half*8.
__device__ __forceinline__ v16bf load_frag16(const unsigned short* base, int ld,
                                             int row, int kbase) {
  int lane = threadIdx.x & 31;
  const unsigned short* rp =
      base + (size_t)(row + (lane & 15)) * ld + kbase + ((lane >> 4) << 3);
  uint4 lo = *(const uint4*)(rp);        // K rel 0..7
  uint4 hi = *(const uint4*)(rp + 16);   // K rel 16..23
  v8u u;
  u[0] = lo.x; u[1] = lo.y; u[2] = lo.z; u[3] = lo.w;
  u[4] = hi.x; u[5] = hi.y; u[6] = hi.z; u[7] = hi.w;
  return __builtin_bit_cast(v16bf, u);
}

// Fragment from an LDS tile with 40-ushort (80B) padded row stride:
// 16B-aligned -> two ds_load_b128 per lane; row bank stride 20 dwords
// (gcd(20,64)=4 -> 16 distinct banks across the 16 rows, conflict-free).
__device__ __forceinline__ v16bf load_frag16_lds16(const unsigned short* base, int row) {
  int lane = threadIdx.x & 31;
  const unsigned short* rp = base + (row + (lane & 15)) * 40 + ((lane >> 4) << 3);
  uint4 lo = *(const uint4*)(rp);
  uint4 hi = *(const uint4*)(rp + 16);
  v8u u;
  u[0] = lo.x; u[1] = lo.y; u[2] = lo.z; u[3] = lo.w;
  u[4] = hi.x; u[5] = hi.y; u[6] = hi.z; u[7] = hi.w;
  return __builtin_bit_cast(v16bf, u);
}

// Same layout, dword loads (attention P staging rows are only 4B aligned).
__device__ __forceinline__ v16bf load_frag16_lds(const unsigned short* base, int ld) {
  int lane = threadIdx.x & 31;
  const unsigned short* rp = base + (lane & 15) * ld + ((lane >> 4) << 3);
  v8u u;
#pragma unroll
  for (int v = 0; v < 8; ++v) {
    int off = ((v & 4) << 2) + ((v & 3) << 1);  // 0,2,4,6,16,18,20,22
    u[v] = *(const unsigned int*)(rp + off);
  }
  return __builtin_bit_cast(v16bf, u);
}

#define WMMA_BF16(A, B, C) \
  __builtin_amdgcn_wmma_f32_16x16x32_bf16(false, (A), false, (B), (short)0, (C), false, false)

// ---------------------------------------------------------------------------
// Elementwise converters
// ---------------------------------------------------------------------------
__global__ void cvt_bf16(const float* __restrict__ in, unsigned short* __restrict__ out,
                         long n) {
  long i = (long)blockIdx.x * 256 + threadIdx.x;
  if (i < n) out[i] = f2bf(in[i]);
}

// in [R,C] f32 -> out [C,R] bf16 (weights stored K-contiguous per output col)
__global__ void cvt_bf16_T(const float* __restrict__ in, unsigned short* __restrict__ out,
                           int R, int C) {
  long i = (long)blockIdx.x * 256 + threadIdx.x;
  long n = (long)R * C;
  if (i < n) {
    int r = (int)(i / C), c = (int)(i % C);
    out[(size_t)c * R + r] = f2bf(in[i]);
  }
}

// ---------------------------------------------------------------------------
// bf16 GEMM:  C[M,N] = A[M,K] * Bt[N,K]^T + bias[N]
// 256 threads = 8 waves (4x2), block tile 128x64, wave tile 32x32.
// Double-buffered async global->LDS staging (3 b128 copies/thread/K-step).
// ---------------------------------------------------------------------------
__global__ __launch_bounds__(256) void gemm_bf16(
    const unsigned short* __restrict__ A, const unsigned short* __restrict__ Bt,
    const float* __restrict__ bias, void* __restrict__ Cout,
    int M, int N, int K, int out_bf16) {
  __shared__ unsigned short As[2][128 * 40];   // 128 rows x 32 bf16, 80B stride
  __shared__ unsigned short Bs[2][64 * 40];    // 64 rows  x 32 bf16, 80B stride

  int tid  = threadIdx.x;
  int wave = tid >> 5;
  int lane = tid & 31;
  int nBlocksN = N >> 6;
  int bm = blockIdx.x / nBlocksN;
  int bn = blockIdx.x % nBlocksN;
  int mblk = bm * 128;
  int nblk = bn * 64;
  int wm = (wave & 3) * 32;          // wave row offset within block tile
  int wn = (wave >> 2) * 32;         // wave col offset within block tile

  // per-thread staging assignment: A chunks {tid, tid+256}, B chunk {tid}
  int ar0 = (tid) >> 2,       ac0 = (tid & 3) * 8;
  int ar1 = (tid + 256) >> 2, ac1 = ac0;
  int br  = tid >> 2,         bc  = ac0;

  v8f acc00 = vzero8(), acc01 = vzero8(), acc10 = vzero8(), acc11 = vzero8();

  const int nk = K >> 5;
  // prologue: stage K-step 0 into buffer 0
  async_copy16(A  + (size_t)(mblk + ar0) * K + ac0, &As[0][ar0 * 40 + ac0]);
  async_copy16(A  + (size_t)(mblk + ar1) * K + ac1, &As[0][ar1 * 40 + ac1]);
  async_copy16(Bt + (size_t)(nblk + br ) * K + bc,  &Bs[0][br  * 40 + bc ]);

  for (int ks = 0; ks < nk; ++ks) {
    int cur = ks & 1;
    if (ks + 1 < nk) {
      int kb = (ks + 1) << 5;
      async_copy16(A  + (size_t)(mblk + ar0) * K + kb + ac0, &As[cur ^ 1][ar0 * 40 + ac0]);
      async_copy16(A  + (size_t)(mblk + ar1) * K + kb + ac1, &As[cur ^ 1][ar1 * 40 + ac1]);
      async_copy16(Bt + (size_t)(nblk + br ) * K + kb + bc,  &Bs[cur ^ 1][br  * 40 + bc ]);
      ASYNC_WAIT(3);     // in-order: current buffer's 3 copies have landed
    } else {
      ASYNC_WAIT(0);
    }
    __syncthreads();     // publish all waves' copies of current buffer

    v16bf a0 = load_frag16_lds16(&As[cur][0], wm);
    v16bf a1 = load_frag16_lds16(&As[cur][0], wm + 16);
    v16bf b0 = load_frag16_lds16(&Bs[cur][0], wn);
    v16bf b1 = load_frag16_lds16(&Bs[cur][0], wn + 16);
    acc00 = WMMA_BF16(a0, b0, acc00);
    acc01 = WMMA_BF16(a0, b1, acc01);
    acc10 = WMMA_BF16(a1, b0, acc10);
    acc11 = WMMA_BF16(a1, b1, acc11);

    __syncthreads();     // reads done before this buffer is overwritten
  }

  int col = lane & 15, half = lane >> 4;
  v8f* accs[4] = {&acc00, &acc01, &acc10, &acc11};
#pragma unroll
  for (int i = 0; i < 2; ++i) {
#pragma unroll
    for (int j = 0; j < 2; ++j) {
      int c = nblk + wn + j * 16 + col;
      float bv = bias[c];
      v8f& a = *accs[i * 2 + j];
#pragma unroll
      for (int r = 0; r < 8; ++r) {
        int row = mblk + wm + i * 16 + r + half * 8;
        float val = a[r] + bv;
        if (out_bf16)
          ((unsigned short*)Cout)[(size_t)row * N + c] = f2bf(val);
        else
          ((float*)Cout)[(size_t)row * N + c] = val;
      }
    }
  }
}

// ---------------------------------------------------------------------------
// Scatter [B*S, 3D] bf16 -> Q,K [B,H,S,64] and V transposed [B,H,64,S]
// ---------------------------------------------------------------------------
__global__ void qkv_scatter(const unsigned short* __restrict__ qkv,
                            unsigned short* __restrict__ Q,
                            unsigned short* __restrict__ Km,
                            unsigned short* __restrict__ Vt) {
  long i = (long)blockIdx.x * 256 + threadIdx.x;
  const long n = (long)BATCH * SEQ * EMB3;
  if (i >= n) return;
  int c   = (int)(i % EMB3);
  long t  = i / EMB3;                 // token = b*SEQ + s
  int which = c >> 10;                // 0=q 1=k 2=v
  int rem   = c & 1023;
  int h = rem >> 6, d = rem & 63;
  int b = (int)(t / SEQ), s = (int)(t % SEQ);
  unsigned short v = qkv[i];
  size_t bh = (size_t)b * NH + h;
  if (which == 0)       Q[(bh * SEQ + s) * HD + d]  = v;
  else if (which == 1)  Km[(bh * SEQ + s) * HD + d] = v;
  else                  Vt[(bh * HD + d) * SEQ + s] = v;
}

// ---------------------------------------------------------------------------
// Causal flash attention: one wave per 16-query tile, 32-key blocks.
// ---------------------------------------------------------------------------
__global__ __launch_bounds__(128) void attn_kernel(
    const unsigned short* __restrict__ Q, const unsigned short* __restrict__ Kmat,
    const unsigned short* __restrict__ Vt, unsigned short* __restrict__ Obf) {
  __shared__ unsigned short pls[4][16][34];   // per-wave P staging (C->A relayout)

  int wave = threadIdx.x >> 5;
  int lane = threadIdx.x & 31;
  int gw = blockIdx.x * 4 + wave;             // global wave id
  int qt = gw & (SEQ / 16 - 1);               // q-tile within (b,h)
  int bh = gw >> 7;                           // 0..B*NH-1
  const unsigned short* Qp = Q    + (size_t)bh * SEQ * HD;
  const unsigned short* Kp = Kmat + (size_t)bh * SEQ * HD;
  const unsigned short* Vp = Vt   + (size_t)bh * HD * SEQ;

  int qrowbase = qt * 16;
  int half = lane >> 4, col = lane & 15;

  v16bf q0 = load_frag16(Qp, HD, qrowbase, 0);
  v16bf q1 = load_frag16(Qp, HD, qrowbase, 32);

  v8f acc[4];
#pragma unroll
  for (int t = 0; t < 4; ++t) acc[t] = vzero8();
  float m[8], l[8];
#pragma unroll
  for (int r = 0; r < 8; ++r) { m[r] = -1e30f; l[r] = 0.0f; }

  int kblocks = (qrowbase + 16 + 31) >> 5;
  for (int j = 0; j < kblocks; ++j) {
    int kb = j << 5;
    // scores: Q(16x64) * K^T(64x32), two 16x16 N-tiles, K split 2x32
    v16bf k00 = load_frag16(Kp, HD, kb,      0);
    v16bf k01 = load_frag16(Kp, HD, kb,      32);
    v16bf k10 = load_frag16(Kp, HD, kb + 16, 0);
    v16bf k11 = load_frag16(Kp, HD, kb + 16, 32);
    v8f s0 = vzero8(), s1 = vzero8();
    s0 = WMMA_BF16(q0, k00, s0);
    s0 = WMMA_BF16(q1, k01, s0);
    s1 = WMMA_BF16(q0, k10, s1);
    s1 = WMMA_BF16(q1, k11, s1);

    float p0[8], p1[8];
#pragma unroll
    for (int r = 0; r < 8; ++r) {
      int qrow = qrowbase + r + half * 8;
      float v0 = s0[r] * 0.125f;              // 1/sqrt(64)
      float v1 = s1[r] * 0.125f;
      if (kb + col > qrow)      v0 = -1e30f;  // causal mask
      if (kb + 16 + col > qrow) v1 = -1e30f;
      p0[r] = v0; p1[r] = v1;
    }
    // online softmax update (row stats: xor-shuffles stay within 16-lane halves)
#pragma unroll
    for (int r = 0; r < 8; ++r) {
      float bmax = fmaxf(p0[r], p1[r]);
#pragma unroll
      for (int st = 8; st >= 1; st >>= 1)
        bmax = fmaxf(bmax, __shfl_xor(bmax, st, 32));
      float mn = fmaxf(m[r], bmax);
      float a  = __expf(m[r] - mn);
      m[r] = mn;
      float e0 = __expf(p0[r] - mn);
      float e1 = __expf(p1[r] - mn);
      p0[r] = e0; p1[r] = e1;
      float rs = e0 + e1;
#pragma unroll
      for (int st = 8; st >= 1; st >>= 1)
        rs += __shfl_xor(rs, st, 32);
      l[r] = l[r] * a + rs;
#pragma unroll
      for (int t = 0; t < 4; ++t) acc[t][r] = acc[t][r] * a;
    }
    // stage P (C layout) into LDS, read back in A layout
#pragma unroll
    for (int r = 0; r < 8; ++r) {
      pls[wave][r + half * 8][col]      = f2bf(p0[r]);
      pls[wave][r + half * 8][col + 16] = f2bf(p1[r]);
    }
    __asm__ volatile("" ::: "memory");   // same-wave DS ops are in-order
    v16bf pa = load_frag16_lds(&pls[wave][0][0], 34);
    // O += P(16x32) * V(32x64): 4 d-tiles
#pragma unroll
    for (int t = 0; t < 4; ++t) {
      v16bf vf = load_frag16(Vp, SEQ, t * 16, kb);
      acc[t] = WMMA_BF16(pa, vf, acc[t]);
    }
  }

  // epilogue: normalize, write bf16 attention output in [B,S,H*HD] order
  int b = bh >> 4, h = bh & 15;
#pragma unroll
  for (int t = 0; t < 4; ++t) {
#pragma unroll
    for (int r = 0; r < 8; ++r) {
      int qrow = qrowbase + r + half * 8;
      float o = acc[t][r] / l[r];
      Obf[((size_t)(b * SEQ + qrow)) * EMB + h * HD + t * 16 + col] = f2bf(o);
    }
  }
}

// ---------------------------------------------------------------------------
// Host launcher
// ---------------------------------------------------------------------------
extern "C" void kernel_launch(void* const* d_in, const int* in_sizes, int n_in,
                              void* d_out, int out_size, void* d_ws, size_t ws_size,
                              hipStream_t stream) {
  const float* x     = (const float*)d_in[0];
  const float* w_qkv = (const float*)d_in[1];
  const float* b_qkv = (const float*)d_in[2];
  const float* w_out = (const float*)d_in[3];
  const float* b_out = (const float*)d_in[4];
  float* out = (float*)d_out;

  char* ws = (char*)d_ws;
  size_t off = 0;
  auto wsalloc = [&](size_t bytes) -> void* {
    void* p = ws + off;
    off = (off + bytes + 255) & ~(size_t)255;
    return p;
  };
  const size_t TOK = (size_t)BATCH * SEQ;                       // 8192
  unsigned short* Xbf   = (unsigned short*)wsalloc(TOK * EMB * 2);
  unsigned short* Wqkvt = (unsigned short*)wsalloc((size_t)EMB * EMB3 * 2);
  unsigned short* Woutt = (unsigned short*)wsalloc((size_t)EMB * EMB * 2);
  unsigned short* QKV   = (unsigned short*)wsalloc(TOK * EMB3 * 2);
  unsigned short* Qm    = (unsigned short*)wsalloc(TOK * EMB * 2);
  unsigned short* Km    = (unsigned short*)wsalloc(TOK * EMB * 2);
  unsigned short* Vt    = (unsigned short*)wsalloc(TOK * EMB * 2);
  unsigned short* Obf   = (unsigned short*)wsalloc(TOK * EMB * 2);

  long nx = (long)TOK * EMB;
  cvt_bf16<<<(unsigned)((nx + 255) / 256), 256, 0, stream>>>(x, Xbf, nx);
  long nw1 = (long)EMB * EMB3;
  cvt_bf16_T<<<(unsigned)((nw1 + 255) / 256), 256, 0, stream>>>(w_qkv, Wqkvt, EMB, EMB3);
  long nw2 = (long)EMB * EMB;
  cvt_bf16_T<<<(unsigned)((nw2 + 255) / 256), 256, 0, stream>>>(w_out, Woutt, EMB, EMB);

  // QKV projection: [8192,1024] x [1024,3072] -> bf16 [8192,3072]
  gemm_bf16<<<(8192 / 128) * (EMB3 / 64), 256, 0, stream>>>(
      Xbf, Wqkvt, b_qkv, QKV, 8192, EMB3, EMB, 1);

  long nq = (long)TOK * EMB3;
  qkv_scatter<<<(unsigned)((nq + 255) / 256), 256, 0, stream>>>(QKV, Qm, Km, Vt);

  // attention: B*H*(S/16) waves, 4 waves per 128-thread block
  attn_kernel<<<(BATCH * NH * (SEQ / 16)) / 4, 128, 0, stream>>>(Qm, Km, Vt, Obf);

  // output projection: [8192,1024] x [1024,1024] + bias -> f32 d_out
  gemm_bf16<<<(8192 / 128) * (EMB / 64), 256, 0, stream>>>(
      Obf, Woutt, b_out, out, 8192, EMB, EMB, 0);
}
